// myModel_2688649527734
// MI455X (gfx1250) — matrix operand order repaired and checked by reading, not compiled, not run on hardware
//
#include <hip/hip_runtime.h>
#include <hip/hip_bf16.h>
#include <math.h>

// ---- model dims ----
#define BB 256
#define NN 96
#define EE 1024
#define HD 300
#define LL 2
#define LH 600
#define NC 7
#define G4 (4 * LH)             // 2400
#define BN (BB * NN)            // 24576
// ---- padded dims (all K multiples of 32, all Nc row-padded to 64) ----
#define SEG 320                 // padded H segment width (HD=300 -> 320)
#define SEQKP (EE + 3 * SEG)    // 1984, padded LSTM input width
#define HDP 320                 // padded HD
#define LHP 608                 // padded LH
#define G4RP 2432               // G4=2400 rows padded to 64
#define HOFF(l) (EE + SEG * (l))

typedef __attribute__((ext_vector_type(16))) _Float16 v16h;
typedef __attribute__((ext_vector_type(8)))  _Float16 v8h;
typedef __attribute__((ext_vector_type(8)))  float    v8f;

__device__ __forceinline__ v16h cat8(v8h a, v8h b) {
  return __builtin_shufflevector(a, b, 0, 1, 2, 3, 4, 5, 6, 7,
                                 8, 9, 10, 11, 12, 13, 14, 15);
}

// A fragment: ap already includes row*lda + (hi?16:0); two aligned b128 loads.
__device__ __forceinline__ v16h frag_a_vec(const _Float16* __restrict__ ap, int k0) {
  return cat8(*(const v8h*)(ap + k0), *(const v8h*)(ap + k0 + 8));
}
// B^T fragment: wp already includes col*ldw + (hi?16:0); two aligned b128 loads.
__device__ __forceinline__ v16h frag_b_vec(const _Float16* __restrict__ wp, int k0) {
  return cat8(*(const v8h*)(wp + k0), *(const v8h*)(wp + k0 + 8));
}

#define WMMA(A_, B_, C_) \
  __builtin_amdgcn_wmma_f32_16x16x32_f16(false, (A_), false, (B_), (short)0, (C_), false, false)

// ============ generic WMMA GEMM: Out = act(A[MxKP] @ W[NcxKP]^T + bias) ============
// One wave computes a 32x64 tile: 2 row tiles x 4 col tiles, double-buffered K loop
// with a peeled (load-free) final iteration. KP is a compile-time constant so load
// addresses fold to base + immediate offsets.
// Requirements: M % 32 == 0; KP % 32 == 0, KP >= 64; A rows have >= KP readable halves
// with zeros in the pad; W has >= ceil64(Nc) rows of ldw >= KP halves, zero K-pad.
template <int KP>
__global__ __launch_bounds__(32)
void gemm_wt_kernel(const _Float16* __restrict__ A, int lda,
                    const _Float16* __restrict__ W, int ldw,
                    const float* __restrict__ bias,
                    float* __restrict__ Out, int ldo,
                    _Float16* __restrict__ Out16, int ldo16,
                    _Float16* __restrict__ OutT16,     // optional per-batch transpose
                    int M, int Nc,
                    int act, const float* __restrict__ aptr) {
  const int lane = threadIdx.x;
  const bool hi = lane >= 16;
  const int r = lane & 15;
  const int row0 = blockIdx.y * 32;
  const int col0 = blockIdx.x * 64;
  const _Float16* ap0 = A + (size_t)(row0 + r) * lda + (hi ? 16 : 0);
  const _Float16* ap1 = ap0 + (size_t)16 * lda;
  const _Float16* wp = W + (size_t)(col0 + r) * ldw + (hi ? 16 : 0);
  const size_t wstp = (size_t)16 * ldw;

  v8f acc[2][4];
#pragma unroll
  for (int rt = 0; rt < 2; ++rt)
#pragma unroll
    for (int t = 0; t < 4; ++t) acc[rt][t] = (v8f){};

  // prologue: current fragments for k0 = 0
  v16h aC0 = frag_a_vec(ap0, 0);
  v16h aC1 = frag_a_vec(ap1, 0);
  v16h bC0 = frag_b_vec(wp, 0);
  v16h bC1 = frag_b_vec(wp + wstp, 0);
  v16h bC2 = frag_b_vec(wp + 2 * wstp, 0);
  v16h bC3 = frag_b_vec(wp + 3 * wstp, 0);

#pragma unroll 4
  for (int k0 = 0; k0 < KP - 32; k0 += 32) {
    const int kn = k0 + 32;
    __builtin_prefetch(ap0 + kn + 32, 0, 1);   // speculative; <=128B over-read is benign
    // issue next-iteration loads BEFORE consuming current fragments
    v16h aN0 = frag_a_vec(ap0, kn);
    v16h aN1 = frag_a_vec(ap1, kn);
    v16h bN0 = frag_b_vec(wp, kn);
    v16h bN1 = frag_b_vec(wp + wstp, kn);
    v16h bN2 = frag_b_vec(wp + 2 * wstp, kn);
    v16h bN3 = frag_b_vec(wp + 3 * wstp, kn);

    acc[0][0] = WMMA(aC0, bC0, acc[0][0]);
    acc[1][0] = WMMA(aC1, bC0, acc[1][0]);
    acc[0][1] = WMMA(aC0, bC1, acc[0][1]);
    acc[1][1] = WMMA(aC1, bC1, acc[1][1]);
    acc[0][2] = WMMA(aC0, bC2, acc[0][2]);
    acc[1][2] = WMMA(aC1, bC2, acc[1][2]);
    acc[0][3] = WMMA(aC0, bC3, acc[0][3]);
    acc[1][3] = WMMA(aC1, bC3, acc[1][3]);

    aC0 = aN0; aC1 = aN1;
    bC0 = bN0; bC1 = bN1; bC2 = bN2; bC3 = bN3;
  }
  // peeled final iteration: no loads
  acc[0][0] = WMMA(aC0, bC0, acc[0][0]);
  acc[1][0] = WMMA(aC1, bC0, acc[1][0]);
  acc[0][1] = WMMA(aC0, bC1, acc[0][1]);
  acc[1][1] = WMMA(aC1, bC1, acc[1][1]);
  acc[0][2] = WMMA(aC0, bC2, acc[0][2]);
  acc[1][2] = WMMA(aC1, bC2, acc[1][2]);
  acc[0][3] = WMMA(aC0, bC3, acc[0][3]);
  acc[1][3] = WMMA(aC1, bC3, acc[1][3]);

  const float pa = act ? *aptr : 0.f;
  const int rbase = hi ? 8 : 0;
#pragma unroll
  for (int rt = 0; rt < 2; ++rt) {
#pragma unroll
    for (int t = 0; t < 4; ++t) {
      int col = col0 + 16 * t + r;
      if (col < Nc) {
        float bval = bias ? bias[col] : 0.f;
#pragma unroll
        for (int i = 0; i < 8; ++i) {
          int rowg = row0 + 16 * rt + rbase + i;
          float v = acc[rt][t][i] + bval;
          if (act) v = (v >= 0.f) ? v : pa * v;
          if (Out)   Out[(size_t)rowg * ldo + col] = v;
          if (Out16) Out16[(size_t)rowg * ldo16 + col] = (_Float16)v;
          if (OutT16) {
            int bb = rowg / NN, nn = rowg - bb * NN;
            OutT16[((size_t)bb * HDP + col) * NN + nn] = (_Float16)v;
          }
        }
      }
    }
  }
}

// ============ adjacency: adj = (a*entro[:,None,:] + (1-a)*f@f^T) > th ============
__global__ __launch_bounds__(32)
void sim_adj_kernel(const _Float16* __restrict__ fh,
                    const float* __restrict__ entro,
                    const float* __restrict__ alphap,
                    const float* __restrict__ threshp,
                    unsigned char* __restrict__ adj) {
  const int lane = threadIdx.x;
  const bool hi = lane >= 16;
  const int r = lane & 15;
  const int bidx = blockIdx.z;
  const int row0 = blockIdx.y * 16;
  const int col0 = blockIdx.x * 16;
  const _Float16* Fb = fh + (size_t)bidx * NN * EE;
  const _Float16* ap = Fb + (size_t)(row0 + r) * EE + (hi ? 16 : 0);
  const _Float16* wp = Fb + (size_t)(col0 + r) * EE + (hi ? 16 : 0);
  v8f acc = {};
  v16h aC = frag_a_vec(ap, 0);
  v16h bC = frag_b_vec(wp, 0);
#pragma unroll 4
  for (int k0 = 0; k0 < EE - 32; k0 += 32) {
    v16h aN = frag_a_vec(ap, k0 + 32);
    v16h bN = frag_b_vec(wp, k0 + 32);
    acc = WMMA(aC, bC, acc);
    aC = aN; bC = bN;
  }
  acc = WMMA(aC, bC, acc);
  const float al = *alphap, th = *threshp;
  const int col = col0 + r;
  const int rbase = hi ? 8 : 0;
  const float ej = al * entro[bidx * NN + col];
#pragma unroll
  for (int i = 0; i < 8; ++i) {
    int rowg = row0 + rbase + i;
    float comb = ej + (1.f - al) * acc[i];
    adj[((size_t)bidx * NN + rowg) * NN + col] = (comb > th) ? 1 : 0;
  }
}

// ============ GAT apply: M = (attn*sm)@V0 + (attn*(1-sm))@V1 (V pre-transposed) ======
__global__ __launch_bounds__(32)
void attn_apply_kernel(const _Float16* __restrict__ asm16,
                       const _Float16* __restrict__ ausm16,
                       const _Float16* __restrict__ V0T,   // [BB][HDP][NN]
                       const _Float16* __restrict__ V1T,
                       _Float16* __restrict__ out16, int ldo16) {
  const int lane = threadIdx.x;
  const bool hi = lane >= 16;
  const int r = lane & 15;
  const int bidx = blockIdx.z;
  const int row0 = blockIdx.y * 16;
  const int col0 = blockIdx.x * 64;
  const _Float16* a1p = asm16 + (size_t)bidx * NN * NN + (size_t)(row0 + r) * NN + (hi ? 16 : 0);
  const _Float16* a2p = ausm16 + (size_t)bidx * NN * NN + (size_t)(row0 + r) * NN + (hi ? 16 : 0);
  const _Float16* w1p = V0T + ((size_t)bidx * HDP + col0 + r) * NN + (hi ? 16 : 0);
  const _Float16* w2p = V1T + ((size_t)bidx * HDP + col0 + r) * NN + (hi ? 16 : 0);
  const size_t wstp = (size_t)16 * NN;
  v8f acc[4] = {{}, {}, {}, {}};
#pragma unroll
  for (int k0 = 0; k0 < NN; k0 += 32) {
    v16h a1 = frag_a_vec(a1p, k0);
    v16h a2 = frag_a_vec(a2p, k0);
    v16h b1[4], b2[4];
#pragma unroll
    for (int t = 0; t < 4; ++t) b1[t] = frag_b_vec(w1p + (size_t)t * wstp, k0);
#pragma unroll
    for (int t = 0; t < 4; ++t) b2[t] = frag_b_vec(w2p + (size_t)t * wstp, k0);
#pragma unroll
    for (int t = 0; t < 4; ++t) {
      acc[t] = WMMA(a1, b1[t], acc[t]);
      acc[t] = WMMA(a2, b2[t], acc[t]);
    }
  }
  const int rbase = hi ? 8 : 0;
#pragma unroll
  for (int t = 0; t < 4; ++t) {
    int col = col0 + 16 * t + r;
    if (col < HD) {
#pragma unroll
      for (int i = 0; i < 8; ++i) {
        int rowg = row0 + rbase + i;
        out16[((size_t)bidx * NN + rowg) * ldo16 + col] = (_Float16)acc[t][i];
      }
    }
  }
}

// ============ normalize features; emit raw-f16 and normalized-f16 ============
__global__ __launch_bounds__(256)
void prep_kernel(const float* __restrict__ feat,
                 _Float16* __restrict__ raw16, _Float16* __restrict__ norm16) {
  __shared__ float red[256];
  int rr = blockIdx.x;
  int tid = threadIdx.x;
  const float* fr = feat + (size_t)rr * EE;
  float ss = 0.f;
  for (int i = tid; i < EE; i += 256) { float v = fr[i]; ss += v * v; }
  red[tid] = ss; __syncthreads();
  for (int s = 128; s > 0; s >>= 1) { if (tid < s) red[tid] += red[tid + s]; __syncthreads(); }
  float inv = 1.f / (sqrtf(red[0]) + 1e-8f);
  for (int i = tid; i < EE; i += 256) {
    float v = fr[i];
    raw16[(size_t)rr * EE + i]  = (_Float16)v;
    norm16[(size_t)rr * EE + i] = (_Float16)(v * inv);
  }
}

// ============ GAT q/k projections ============
__global__ void qk_kernel(const _Float16* __restrict__ Hl, int lds,
                          const float* __restrict__ gw,
                          float* __restrict__ q, float* __restrict__ k) {
  int t = blockIdx.x * blockDim.x + threadIdx.x;
  if (t >= BN) return;
  const _Float16* h = Hl + (size_t)t * lds;
  float sq = 0.f, sk = 0.f;
  for (int d = 0; d < HD; ++d) {
    float hv = (float)h[d];
    sq += hv * gw[d];
    sk += hv * gw[HD + d];
  }
  q[t] = sq; k[t] = sk;
}

// ============ masked softmax over N=96, wave per row ============
__global__ __launch_bounds__(32)
void attn_softmax_kernel(const float* __restrict__ q, const float* __restrict__ k,
                         const float* __restrict__ gbp,
                         const unsigned char* __restrict__ adj,
                         const int* __restrict__ s_mask,
                         _Float16* __restrict__ asm16, _Float16* __restrict__ ausm16) {
  int row = blockIdx.x;
  int bidx = row / NN;
  int lane = threadIdx.x;
  float qv = q[row];
  float gb = *gbp;
  const unsigned char* arow = adj + (size_t)row * NN;
  float s[3];
#pragma unroll
  for (int m = 0; m < 3; ++m) {
    int j = lane + 32 * m;
    float sc = qv + k[bidx * NN + j] + gb;
    s[m] = arow[j] ? sc : -1e9f;
  }
  float mx = fmaxf(fmaxf(s[0], s[1]), s[2]);
  for (int o = 16; o > 0; o >>= 1) mx = fmaxf(mx, __shfl_xor(mx, o, 32));
  float e[3], sum = 0.f;
#pragma unroll
  for (int m = 0; m < 3; ++m) { e[m] = __expf(s[m] - mx); sum += e[m]; }
  for (int o = 16; o > 0; o >>= 1) sum += __shfl_xor(sum, o, 32);
  float inv = 1.f / sum;
  const int* smrow = s_mask + (size_t)row * NN;
#pragma unroll
  for (int m = 0; m < 3; ++m) {
    int j = lane + 32 * m;
    float a = e[m] * inv;
    float sv = (float)smrow[j];
    asm16[(size_t)row * NN + j]  = (_Float16)(a * sv);
    ausm16[(size_t)row * NN + j] = (_Float16)(a * (1.f - sv));
  }
}

// ============ LSTM helpers ============
__global__ void bias2_kernel(const float* a, const float* b, float* o) {
  int i = blockIdx.x * blockDim.x + threadIdx.x;
  if (i < G4) o[i] = a[i] + b[i];
}

__global__ void lstm_cell_kernel(const float* __restrict__ zin, int t,
                                 const float* __restrict__ gtmp,
                                 float* __restrict__ h, float* __restrict__ c,
                                 _Float16* __restrict__ h16, _Float16* __restrict__ hs16) {
  int u = blockIdx.x * blockDim.x + threadIdx.x;
  if (u >= BB * LH) return;
  int b = u / LH, j = u - b * LH;
  const float* z = zin + ((size_t)b * NN + t) * G4;
  const float* g = gtmp + (size_t)b * G4;
  float gi = z[j] + g[j];
  float gf = z[LH + j] + g[LH + j];
  float gg = z[2 * LH + j] + g[2 * LH + j];
  float go = z[3 * LH + j] + g[3 * LH + j];
  float si = 1.f / (1.f + __expf(-gi));
  float sf = 1.f / (1.f + __expf(-gf));
  float so = 1.f / (1.f + __expf(-go));
  float cn = sf * c[u] + si * tanhf(gg);
  float hn = so * tanhf(cn);
  c[u] = cn; h[u] = hn;
  h16[(size_t)b * LHP + j] = (_Float16)hn;
  hs16[((size_t)b * NN + t) * LHP + j] = (_Float16)hn;
}

// ============ padding weight converters ============
__global__ void cvt_pad_kernel(const float* __restrict__ src, _Float16* __restrict__ dst,
                               int rows, int cols, int rowsP, int ldP) {
  int i = blockIdx.x * blockDim.x + threadIdx.x;
  if (i >= rowsP * ldP) return;
  int rr = i / ldP, cc = i - rr * ldP;
  float v = (rr < rows && cc < cols) ? src[(size_t)rr * cols + cc] : 0.f;
  dst[i] = (_Float16)v;
}

// w_ih (2400 x 1924) -> padded (2432 x 1984) with segmented column remap matching seq.
__global__ void cvt_wih_kernel(const float* __restrict__ src, _Float16* __restrict__ dst) {
  int i = blockIdx.x * blockDim.x + threadIdx.x;
  if (i >= G4RP * SEQKP) return;
  int rr = i / SEQKP, cc = i - rr * SEQKP;
  float v = 0.f;
  if (rr < G4) {
    if (cc < EE) {
      v = src[(size_t)rr * (EE + 3 * HD) + cc];
    } else {
      int seg = (cc - EE) / SEG, pos = (cc - EE) - seg * SEG;
      if (pos < HD) v = src[(size_t)rr * (EE + 3 * HD) + EE + seg * HD + pos];
    }
  }
  dst[i] = (_Float16)v;
}

// ============ host orchestration ============
static inline dim3 tiles(int m, int n) { return dim3((n + 63) / 64, m / 32); }

extern "C" void kernel_launch(void* const* d_in, const int* in_sizes, int n_in,
                              void* d_out, int out_size, void* d_ws, size_t ws_size,
                              hipStream_t stream) {
  const float* features = (const float*)d_in[0];
  const int*   s_mask   = (const int*)d_in[1];
  const float* entro    = (const float*)d_in[2];
  const float* alpha    = (const float*)d_in[3];
  const float* thresh   = (const float*)d_in[4];
  const float* fc1_w    = (const float*)d_in[5];
  const float* fc1_b    = (const float*)d_in[6];
  const float* gat_w    = (const float*)d_in[7];
  const float* gat_b    = (const float*)d_in[8];
  const float* wr0      = (const float*)d_in[9];
  const float* wr1      = (const float*)d_in[10];
  const float* enh_w    = (const float*)d_in[11];
  const float* enh_b    = (const float*)d_in[12];
  const float* prelu_a  = (const float*)d_in[13];
  const float* w_ih     = (const float*)d_in[14];
  const float* w_hh     = (const float*)d_in[15];
  const float* b_ih     = (const float*)d_in[16];
  const float* b_hh     = (const float*)d_in[17];
  const float* mlp_w0   = (const float*)d_in[18];
  const float* mlp_b0   = (const float*)d_in[19];
  const float* mlp_a0   = (const float*)d_in[20];
  const float* mlp_w1   = (const float*)d_in[21];
  const float* mlp_b1   = (const float*)d_in[22];
  const float* mlp_a1   = (const float*)d_in[23];
  const float* out_w    = (const float*)d_in[24];
  const float* out_b    = (const float*)d_in[25];
  float* out = (float*)d_out;

  // ---- workspace carve (all 256-B aligned) ----
  char* base = (char*)d_ws;
  size_t off = 0;
  auto alloc = [&](size_t bytes) -> char* {
    char* p = base + off;
    off = (off + bytes + 255) & ~(size_t)255;
    return p;
  };
  _Float16* fraw16 = (_Float16*)alloc((size_t)BN * EE * 2);
  _Float16* fh16   = (_Float16*)alloc((size_t)BN * EE * 2);
  _Float16* seq16  = (_Float16*)alloc((size_t)BN * SEQKP * 2);
  unsigned char* adj = (unsigned char*)alloc((size_t)BN * NN);
  _Float16* fc1w16 = (_Float16*)alloc((size_t)320 * EE * 2);
  _Float16* enh16  = (_Float16*)alloc((size_t)EE * EE * 2);
  _Float16* wr016  = (_Float16*)alloc((size_t)LL * 320 * 320 * 2);
  _Float16* wr116  = (_Float16*)alloc((size_t)LL * 320 * 320 * 2);
  _Float16* wih16  = (_Float16*)alloc((size_t)G4RP * SEQKP * 2);
  _Float16* whh16  = (_Float16*)alloc((size_t)G4RP * LHP * 2);
  _Float16* mw016  = (_Float16*)alloc((size_t)320 * LHP * 2);
  _Float16* mw116  = (_Float16*)alloc((size_t)320 * 320 * 2);
  _Float16* outw16 = (_Float16*)alloc((size_t)64 * 320 * 2);
  float* qb        = (float*)alloc((size_t)BN * 4);
  float* kb        = (float*)alloc((size_t)BN * 4);
  _Float16* asm16  = (_Float16*)alloc((size_t)BN * NN * 2);
  _Float16* ausm16 = (_Float16*)alloc((size_t)BN * NN * 2);
  _Float16* V0T    = (_Float16*)alloc((size_t)BB * HDP * NN * 2);
  _Float16* V1T    = (_Float16*)alloc((size_t)BB * HDP * NN * 2);
  float* zin       = (float*)alloc((size_t)BN * G4 * 4);
  float* gtmp      = (float*)alloc((size_t)BB * G4 * 4);
  float* hbuf      = (float*)alloc((size_t)BB * LH * 4);
  float* cbuf      = (float*)alloc((size_t)BB * LH * 4);
  _Float16* h16    = (_Float16*)alloc((size_t)BB * LHP * 2);
  _Float16* hs16   = (_Float16*)alloc((size_t)BN * LHP * 2);
  _Float16* x116   = (_Float16*)alloc((size_t)BN * HDP * 2);
  _Float16* x216   = (_Float16*)alloc((size_t)BN * HDP * 2);
  float* bias2     = (float*)alloc((size_t)G4 * 4);

  // ---- zero the K-padded activation buffers (pads must stay zero) ----
  hipMemsetAsync(seq16, 0, (size_t)BN * SEQKP * 2, stream);
  hipMemsetAsync(h16, 0, (size_t)BB * LHP * 2, stream);
  hipMemsetAsync(hs16, 0, (size_t)BN * LHP * 2, stream);
  hipMemsetAsync(x116, 0, (size_t)BN * HDP * 2, stream);
  hipMemsetAsync(x216, 0, (size_t)BN * HDP * 2, stream);
  hipMemsetAsync(hbuf, 0, (size_t)BB * LH * 4, stream);
  hipMemsetAsync(cbuf, 0, (size_t)BB * LH * 4, stream);

  // ---- weight conversion with padding ----
  auto cvt = [&](const float* src, _Float16* dst, int rows, int cols, int rowsP, int ldP) {
    int n = rowsP * ldP;
    cvt_pad_kernel<<<(n + 255) / 256, 256, 0, stream>>>(src, dst, rows, cols, rowsP, ldP);
  };
  cvt(fc1_w, fc1w16, HD, EE, 320, EE);
  cvt(enh_w, enh16, EE, EE, EE, EE);
  for (int l = 0; l < LL; ++l) {
    cvt(wr0 + (size_t)l * HD * HD, wr016 + (size_t)l * 320 * 320, HD, HD, 320, 320);
    cvt(wr1 + (size_t)l * HD * HD, wr116 + (size_t)l * 320 * 320, HD, HD, 320, 320);
  }
  { int n = G4RP * SEQKP;
    cvt_wih_kernel<<<(n + 255) / 256, 256, 0, stream>>>(w_ih, wih16); }
  cvt(w_hh, whh16, G4, LH, G4RP, LHP);
  cvt(mlp_w0, mw016, HD, LH, 320, LHP);
  cvt(mlp_w1, mw116, HD, HD, 320, 320);
  cvt(out_w, outw16, NC, HD, 64, 320);

  // ---- feature prep ----
  prep_kernel<<<BN, 256, 0, stream>>>(features, fraw16, fh16);

  // ---- adjacency ----
  sim_adj_kernel<<<dim3(NN / 16, NN / 16, BB), 32, 0, stream>>>(fh16, entro, alpha, thresh, adj);

  // ---- H0 = prelu(features @ fc1_w^T + b) -> seq segment 0 ----
  gemm_wt_kernel<EE><<<tiles(BN, HDP), 32, 0, stream>>>(
      fraw16, EE, fc1w16, EE, fc1_b, nullptr, 0, seq16 + HOFF(0), SEQKP, nullptr,
      BN, HD, 1, prelu_a);

  // ---- feat = prelu(features @ enhance_w^T + b) -> seq[:, 0:E] ----
  gemm_wt_kernel<EE><<<tiles(BN, EE), 32, 0, stream>>>(
      fraw16, EE, enh16, EE, enh_b, nullptr, 0, seq16, SEQKP, nullptr,
      BN, EE, 1, prelu_a);

  // ---- GAT layers ----
  for (int l = 0; l < LL; ++l) {
    const _Float16* Hl = seq16 + HOFF(l);     // lda = SEQKP, zero K-pad in [300,320)
    qk_kernel<<<(BN + 255) / 256, 256, 0, stream>>>(Hl, SEQKP, gat_w + (size_t)l * 2 * HD, qb, kb);
    attn_softmax_kernel<<<BN, 32, 0, stream>>>(qb, kb, gat_b + l, adj, s_mask, asm16, ausm16);
    gemm_wt_kernel<SEG><<<tiles(BN, HDP), 32, 0, stream>>>(
        Hl, SEQKP, wr016 + (size_t)l * 320 * 320, 320, nullptr,
        nullptr, 0, nullptr, 0, V0T, BN, HD, 0, nullptr);
    gemm_wt_kernel<SEG><<<tiles(BN, HDP), 32, 0, stream>>>(
        Hl, SEQKP, wr116 + (size_t)l * 320 * 320, 320, nullptr,
        nullptr, 0, nullptr, 0, V1T, BN, HD, 0, nullptr);
    attn_apply_kernel<<<dim3(HDP / 64, NN / 16, BB), 32, 0, stream>>>(
        asm16, ausm16, V0T, V1T, seq16 + HOFF(l + 1), SEQKP);
  }

  // ---- zin = seq @ w_ih^T + (b_ih + b_hh) ----
  bias2_kernel<<<(G4 + 255) / 256, 256, 0, stream>>>(b_ih, b_hh, bias2);
  gemm_wt_kernel<SEQKP><<<tiles(BN, G4RP), 32, 0, stream>>>(
      seq16, SEQKP, wih16, SEQKP, bias2, zin, G4, nullptr, 0, nullptr,
      BN, G4, 0, nullptr);

  // ---- LSTM recurrence ----
  for (int t = 0; t < NN; ++t) {
    gemm_wt_kernel<LHP><<<tiles(BB, G4RP), 32, 0, stream>>>(
        h16, LHP, whh16, LHP, nullptr, gtmp, G4, nullptr, 0, nullptr,
        BB, G4, 0, nullptr);
    lstm_cell_kernel<<<(BB * LH + 255) / 256, 256, 0, stream>>>(
        zin, t, gtmp, hbuf, cbuf, h16, hs16);
  }

  // ---- MLP head ----
  gemm_wt_kernel<LHP><<<tiles(BN, HDP), 32, 0, stream>>>(
      hs16, LHP, mw016, LHP, mlp_b0, nullptr, 0, x116, HDP, nullptr,
      BN, HD, 1, mlp_a0);
  gemm_wt_kernel<HDP><<<tiles(BN, HDP), 32, 0, stream>>>(
      x116, HDP, mw116, 320, mlp_b1, nullptr, 0, x216, HDP, nullptr,
      BN, HD, 1, mlp_a1);
  gemm_wt_kernel<HDP><<<tiles(BN, 64), 32, 0, stream>>>(
      x216, HDP, outw16, 320, out_b, out, NC, nullptr, 0, nullptr,
      BN, NC, 0, nullptr);

  (void)in_sizes; (void)n_in; (void)out_size; (void)ws_size;
}
